// GPS_4389456577065
// MI455X (gfx1250) — matrix operand order repaired and checked by reading, not compiled, not run on hardware
//
#include <hip/hip_runtime.h>
#include <hip/hip_bf16.h>
#include <math.h>

// ---------------------------------------------------------------------------
// CDNA5 (gfx1250) WMMA types and helpers
// ---------------------------------------------------------------------------
typedef __attribute__((ext_vector_type(16))) __bf16 v16bf;
typedef __attribute__((ext_vector_type(8)))  float  v8f;

union Frag16 { v16bf v; unsigned short s[16]; unsigned int u[8]; };

__device__ __forceinline__ unsigned short f2bf(float f) {
    union { float f; unsigned int u; } c; c.f = f;
    return (unsigned short)((c.u + 0x7FFFu + ((c.u >> 16) & 1u)) >> 16);
}

__device__ __forceinline__ v8f wmma_bf16(v16bf a, v16bf b, v8f c) {
    // D = A(16x32 bf16) * B(32x16 bf16) + C(16x16 f32)
    return __builtin_amdgcn_wmma_f32_16x16x32_bf16(
        false, a, false, b, (short)0, c, false, false);
}

// A-fragment gather from LDS, bf16 row-major [m][k], stride ld (even).
// Layout (16-bit A 16x32): lane half adds K+16; contiguous 16 K
// (VGPR0-3 = K {0..7}, VGPR4-7 = K {8..15}, relative to half offset).
__device__ __forceinline__ v16bf ldsFragA(const unsigned short* base, int ld,
                                          int mbase, int kbase) {
    int lane = threadIdx.x & 31;
    int m = mbase + (lane & 15);
    int koff = kbase + ((lane >> 4) << 4);
    const unsigned int* row = (const unsigned int*)(base + m * ld + koff);
    Frag16 f;
#pragma unroll
    for (int g = 0; g < 4; ++g) {
        f.u[g]     = row[g];       // K = koff + 2g, 2g+1
        f.u[g + 4] = row[g + 4];   // K = koff + 8 + 2g, +1
    }
    return f.v;
}

// B-fragment gather from LDS, bf16 stored as Bt[n][k], stride ld (even).
// Layout used here (16-bit B 32x16, same K order as ldsFragA): lane half adds K+16; contiguous 16 K.
__device__ __forceinline__ v16bf ldsFragB(const unsigned short* base, int ld,
                                          int nbase, int kbase) {
    int lane = threadIdx.x & 31;
    int n = nbase + (lane & 15);
    int k0 = kbase + ((lane >> 4) << 4);
    const unsigned int* row = (const unsigned int*)(base + n * ld + k0);
    Frag16 f;
#pragma unroll
    for (int g = 0; g < 8; ++g) f.u[g] = row[g];
    return f.v;
}

// A-fragment gathered straight from global f32 [m][k] (row stride lda),
// converted to bf16 in-register.
__device__ __forceinline__ v16bf glbFragA(const float* A, int lda,
                                          int mbase, int kbase) {
    int lane = threadIdx.x & 31;
    int m = mbase + (lane & 15);
    const float* row = A + (size_t)m * lda + kbase + ((lane >> 4) << 4);
    Frag16 f;
#pragma unroll
    for (int j = 0; j < 16; ++j) {
        int k = j;
        f.s[j] = f2bf(row[k]);
    }
    return f.v;
}

// B-fragment gathered from global f32 weight W stored [k][n] (stride ldw).
__device__ __forceinline__ v16bf glbFragB(const float* W, int ldw,
                                          int kbase, int nbase) {
    int lane = threadIdx.x & 31;
    int n = nbase + (lane & 15);
    int k0 = kbase + ((lane >> 4) << 4);
    Frag16 f;
#pragma unroll
    for (int j = 0; j < 16; ++j) {
        f.s[j] = f2bf(W[(size_t)(k0 + j) * ldw + n]);
    }
    return f.v;
}

// ---------------------------------------------------------------------------
// WMMA GEMM: C[M,N] = act(A[M,K] @ W[K,N] + bias (+ resid))
// Block = 256 threads (8 wave32). Block tile 128x64, wave tile 32x32
// (4 accumulators, 2x fragment reuse), K-step 32, double-buffered LDS:
// one barrier per K-step, staging of step s+1 overlaps WMMAs of step s.
// ---------------------------------------------------------------------------
template <bool RELU, bool RESID>
__global__ __launch_bounds__(256) void gemm_kernel(
    const float* __restrict__ A, const float* __restrict__ W,
    const float* __restrict__ bias, const float* __restrict__ resid,
    float* __restrict__ C, int M, int N, int K) {
    __shared__ unsigned short As[2][128][40];   // [m][k] bf16
    __shared__ unsigned short Bs[2][64][40];    // [n][k] bf16 (transposed)

    const int tid  = threadIdx.x;
    const int wave = tid >> 5;
    const int lane = tid & 31;
    const int m0 = blockIdx.x * 128;
    const int n0 = blockIdx.y * 64;

    const int wm = (wave & 3) * 32;   // wave rows: 0/32/64/96
    const int wn = (wave >> 2) * 32;  // wave cols: 0/32

    v8f acc[2][2] = {};

    auto stage = [&](int buf, int k0) {
        // A tile 128x32: 2 threads per row, 16 elems each
        int row = tid >> 1;
        int kc  = (tid & 1) * 16;
        const float* ap = A + (size_t)(m0 + row) * K + k0 + kc;
#pragma unroll
        for (int i = 0; i < 16; ++i) As[buf][row][kc + i] = f2bf(ap[i]);
        // B tile 32x64 transposed -> Bs[n][k]
        int kk = tid >> 3;
        int nc = (tid & 7) * 8;
        const float* wp = W + (size_t)(k0 + kk) * N + n0 + nc;
#pragma unroll
        for (int i = 0; i < 8; ++i) Bs[buf][nc + i][kk] = f2bf(wp[i]);
    };

    stage(0, 0);
    __syncthreads();

    const int nsteps = K >> 5;
    for (int s = 0; s < nsteps; ++s) {
        int buf = s & 1;
        if (s + 1 < nsteps) stage(buf ^ 1, (s + 1) << 5);

        v16bf a0 = ldsFragA(&As[buf][0][0], 40, wm, 0);
        v16bf a1 = ldsFragA(&As[buf][0][0], 40, wm + 16, 0);
        v16bf b0 = ldsFragB(&Bs[buf][0][0], 40, wn, 0);
        v16bf b1 = ldsFragB(&Bs[buf][0][0], 40, wn + 16, 0);
        acc[0][0] = wmma_bf16(a0, b0, acc[0][0]);
        acc[0][1] = wmma_bf16(a0, b1, acc[0][1]);
        acc[1][0] = wmma_bf16(a1, b0, acc[1][0]);
        acc[1][1] = wmma_bf16(a1, b1, acc[1][1]);
        __syncthreads();
    }

    // epilogue: C/D layout — col = lane%16, row = v + 8*(lane/16)
#pragma unroll
    for (int im = 0; im < 2; ++im) {
        int rb = m0 + wm + im * 16 + ((lane >> 4) << 3);
#pragma unroll
        for (int in = 0; in < 2; ++in) {
            int col = n0 + wn + in * 16 + (lane & 15);
            float bv = bias[col];
#pragma unroll
            for (int v = 0; v < 8; ++v) {
                int r = rb + v;
                float xv = acc[im][in][v] + bv;
                if (RESID) xv += resid[(size_t)r * N + col];
                if (RELU)  xv = fmaxf(xv, 0.0f);
                C[(size_t)r * N + col] = xv;
            }
        }
    }
}

// ---------------------------------------------------------------------------
// Fused per-(graph, head) attention: computes q/k/v by WMMA straight from h
// and the attn_in weight slice (no qkv materialization), 64x64 softmax in
// LDS, then att @ v^T by WMMA. Grid = (B, HEADS), block = 256 threads.
// ---------------------------------------------------------------------------
__global__ __launch_bounds__(256) void attn_kernel(
    const float* __restrict__ h,      // N x 256
    const float* __restrict__ Wqkv,   // 256 x 768
    const float* __restrict__ bqkv,   // 768
    float* __restrict__ O,            // N x 256 (head hd -> cols hd*64..)
    float scale) {
    __shared__ unsigned short q_s[64][72];   // [q][d] bf16
    __shared__ unsigned short k_s[64][72];   // [key][d] bf16
    __shared__ unsigned short vT_s[64][72];  // [d][key] bf16
    __shared__ unsigned short att_s[64][72]; // [q][key] bf16
    __shared__ float sc[64][68];             // scores f32

    const int tid  = threadIdx.x;
    const int wave = tid >> 5;
    const int lane = tid & 31;
    const int b  = blockIdx.x;
    const int hd = blockIdx.y;
    const int rowBase = b * 64;

    // ---- phase 1: q,k,v = h_rows[64x256] @ W slice[256x64] (+bias) ----
    // mat is fully unrolled so the q/k/v LDS-store selection is resolved at
    // compile time (direct ds_store, no per-element branch ladder).
#pragma unroll
    for (int mat = 0; mat < 3; ++mat) {
        int coff = mat * 256 + hd * 64;
#pragma unroll
        for (int i = 0; i < 2; ++i) {
            int t  = wave * 2 + i;
            int tm = (t >> 2) * 16;
            int tn = (t & 3) * 16;
            v8f acc = {};
            for (int k0 = 0; k0 < 256; k0 += 32) {
                v16bf a  = glbFragA(h, 256, rowBase + tm, k0);
                v16bf bf = glbFragB(Wqkv, 768, k0, coff + tn);
                acc = wmma_bf16(a, bf, acc);
            }
            int n  = tn + (lane & 15);
            int mb = tm + ((lane >> 4) << 3);
            float bv = bqkv[coff + n];
#pragma unroll
            for (int v = 0; v < 8; ++v) {
                int m = mb + v;
                unsigned short bfv = f2bf(acc[v] + bv);
                if (mat == 0)      q_s[m][n]  = bfv;
                else if (mat == 1) k_s[m][n]  = bfv;
                else               vT_s[n][m] = bfv;   // store v transposed
            }
        }
    }
    __syncthreads();

    // ---- phase 2: scores = q @ k^T * scale ----
#pragma unroll
    for (int i = 0; i < 2; ++i) {
        int t  = wave * 2 + i;
        int tm = (t >> 2) * 16;
        int tn = (t & 3) * 16;
        v8f acc = {};
#pragma unroll
        for (int k0 = 0; k0 < 64; k0 += 32) {
            v16bf a  = ldsFragA(&q_s[0][0], 72, tm, k0);
            v16bf bf = ldsFragB(&k_s[0][0], 72, tn, k0);
            acc = wmma_bf16(a, bf, acc);
        }
        int n  = tn + (lane & 15);
        int mb = tm + ((lane >> 4) << 3);
#pragma unroll
        for (int v = 0; v < 8; ++v) sc[mb + v][n] = acc[v] * scale;
    }
    __syncthreads();

    // ---- phase 3: softmax rows (64 rows, one thread each) ----
    if (tid < 64) {
        float mx = -1e30f;
        for (int j = 0; j < 64; ++j) mx = fmaxf(mx, sc[tid][j]);
        float s = 0.0f;
        for (int j = 0; j < 64; ++j) {
            float e = __expf(sc[tid][j] - mx);
            sc[tid][j] = e; s += e;
        }
        float inv = 1.0f / s;
        for (int j = 0; j < 64; ++j) att_s[tid][j] = f2bf(sc[tid][j] * inv);
    }
    __syncthreads();

    // ---- phase 4: o = att @ v  (B operand = vT_s[d][key]) ----
#pragma unroll
    for (int i = 0; i < 2; ++i) {
        int t  = wave * 2 + i;
        int tm = (t >> 2) * 16;
        int tn = (t & 3) * 16;
        v8f acc = {};
#pragma unroll
        for (int k0 = 0; k0 < 64; k0 += 32) {
            v16bf a  = ldsFragA(&att_s[0][0], 72, tm, k0);
            v16bf bf = ldsFragB(&vT_s[0][0], 72, tn, k0);
            acc = wmma_bf16(a, bf, acc);
        }
        int n  = tn + (lane & 15);
        int mb = tm + ((lane >> 4) << 3);
#pragma unroll
        for (int v = 0; v < 8; ++v)
            O[(size_t)(rowBase + mb + v) * 256 + hd * 64 + n] = acc[v];
    }
}

// ---------------------------------------------------------------------------
// Elementwise / reduction utility kernels
// ---------------------------------------------------------------------------
__global__ void zero_kernel(float* p, int n) {
    int i = blockIdx.x * blockDim.x + threadIdx.x;
    if (i < n) p[i] = 0.0f;
}

__global__ void add_kernel(const float* __restrict__ a,
                           const float* __restrict__ b,
                           float* __restrict__ y, int n) {
    int i = blockIdx.x * blockDim.x + threadIdx.x;
    if (i < n) y[i] = a[i] + b[i];
}

// GINE message + scatter: agg[dst] += relu(h[src] + edge_emb[attr])
__global__ __launch_bounds__(256) void scatter_kernel(
    const float* __restrict__ h, const float* __restrict__ edge_emb,
    const int* __restrict__ ei, const int* __restrict__ ea,
    float* __restrict__ agg, int E) {
    int t = blockIdx.x * 256 + threadIdx.x;
    int e = t >> 6;
    if (e >= E) return;
    int c = (t & 63) * 4;
    int src = ei[e], dst = ei[E + e], attr = ea[e];
    const float4 hv = *(const float4*)(h + (size_t)src * 256 + c);
    const float4 ev = *(const float4*)(edge_emb + (size_t)attr * 256 + c);
    float* base = agg + (size_t)dst * 256 + c;
    atomicAdd(base + 0, fmaxf(hv.x + ev.x, 0.0f));
    atomicAdd(base + 1, fmaxf(hv.y + ev.y, 0.0f));
    atomicAdd(base + 2, fmaxf(hv.z + ev.z, 0.0f));
    atomicAdd(base + 3, fmaxf(hv.w + ev.w, 0.0f));
}

// BatchNorm pass 1: per-channel sum / sumsq partials (stats must be zeroed)
__global__ __launch_bounds__(256) void bnstats_kernel(
    const float* __restrict__ x, int Nrows, int C, float* __restrict__ stats,
    int rowsPerBlock) {
    int r0 = blockIdx.x * rowsPerBlock;
    for (int c = threadIdx.x; c < C; c += blockDim.x) {
        float s = 0.0f, ss = 0.0f;
        for (int r = 0; r < rowsPerBlock; ++r) {
            float v = x[(size_t)(r0 + r) * C + c];
            s += v; ss += v * v;
        }
        atomicAdd(&stats[c], s);
        atomicAdd(&stats[C + c], ss);
    }
}

// BatchNorm pass 2: y = (x - mu) * rsqrt(var + eps) * g + b
__global__ void bnapply_kernel(const float* __restrict__ x,
                               float* __restrict__ y,
                               const float* __restrict__ stats,
                               const float* __restrict__ g,
                               const float* __restrict__ b,
                               int Nrows, int C) {
    int i = blockIdx.x * blockDim.x + threadIdx.x;
    int total = Nrows * C;
    if (i >= total) return;
    int c = i % C;
    float inv = 1.0f / (float)Nrows;
    float mu  = stats[c] * inv;
    float var = stats[C + c] * inv - mu * mu;
    y[i] = (x[i] - mu) * rsqrtf(var + 1e-5f) * g[c] + b[c];
}

// Build h = concat(node_emb[x], BN(pe) @ pe_lin_w + pe_lin_b)
__global__ __launch_bounds__(256) void buildh_kernel(
    const int* __restrict__ x, const float* __restrict__ pe,
    const float* __restrict__ stats20, const float* __restrict__ png,
    const float* __restrict__ pnb, const float* __restrict__ node_emb,
    const float* __restrict__ plw, const float* __restrict__ plb,
    float* __restrict__ h, int Nrows) {
    int i = blockIdx.x;
    int c = threadIdx.x;
    if (c < 192) {
        h[(size_t)i * 256 + c] = node_emb[(size_t)x[i] * 192 + c];
    } else {
        int j = c - 192;
        float inv = 1.0f / (float)Nrows;
        float acc = plb[j];
        for (int f = 0; f < 20; ++f) {
            float mu  = stats20[f] * inv;
            float var = stats20[20 + f] * inv - mu * mu;
            float xn = (pe[(size_t)i * 20 + f] - mu) * rsqrtf(var + 1e-5f)
                       * png[f] + pnb[f];
            acc += xn * plw[f * 64 + j];
        }
        h[(size_t)i * 256 + c] = acc;
    }
}

// Per-graph sum over NPG nodes
__global__ void gsum_kernel(const float* __restrict__ h, float* __restrict__ g,
                            int NPG, int C) {
    int b = blockIdx.x;
    int c = threadIdx.x;
    float s = 0.0f;
    for (int i = 0; i < NPG; ++i) s += h[((size_t)b * NPG + i) * C + c];
    g[(size_t)b * C + c] = s;
}

// Final head: out[i] = g2[i,:] @ w + b
__global__ void headfinal_kernel(const float* __restrict__ g2,
                                 const float* __restrict__ w,
                                 const float* __restrict__ b,
                                 float* __restrict__ out, int Bsz, int K) {
    int i = blockIdx.x * blockDim.x + threadIdx.x;
    if (i >= Bsz) return;
    float s = b[0];
    for (int k = 0; k < K; ++k) s += g2[(size_t)i * K + k] * w[k];
    out[i] = s;
}

// ---------------------------------------------------------------------------
// Orchestration
// ---------------------------------------------------------------------------
extern "C" void kernel_launch(void* const* d_in, const int* in_sizes, int n_in,
                              void* d_out, int out_size, void* d_ws,
                              size_t ws_size, hipStream_t stream) {
    const int N = 32768, C = 256, Bg = 512, NPG = 64, E = 262144, L = 4;
    const size_t NC = (size_t)N * C;

    const int*   x        = (const int*)  d_in[0];
    const float* pe       = (const float*)d_in[1];
    const int*   ei       = (const int*)  d_in[2];
    const int*   ea       = (const int*)  d_in[3];
    const float* node_emb = (const float*)d_in[5];
    const float* plw      = (const float*)d_in[6];
    const float* plb      = (const float*)d_in[7];
    const float* png      = (const float*)d_in[8];
    const float* pnb      = (const float*)d_in[9];
    const float* edge_emb = (const float*)d_in[10];
    const float* gw1      = (const float*)d_in[11];
    const float* gb1      = (const float*)d_in[12];
    const float* gw2      = (const float*)d_in[13];
    const float* gb2      = (const float*)d_in[14];
    const float* aiw      = (const float*)d_in[15];
    const float* aib      = (const float*)d_in[16];
    const float* aow      = (const float*)d_in[17];
    const float* aob      = (const float*)d_in[18];
    const float* ng       = (const float*)d_in[19];
    const float* nb       = (const float*)d_in[20];
    const float* m1w      = (const float*)d_in[21];
    const float* m1b      = (const float*)d_in[22];
    const float* m2w      = (const float*)d_in[23];
    const float* m2b      = (const float*)d_in[24];
    const float* hw1      = (const float*)d_in[25];
    const float* hb1      = (const float*)d_in[26];
    const float* hw2      = (const float*)d_in[27];
    const float* hb2      = (const float*)d_in[28];
    const float* hw3      = (const float*)d_in[29];
    const float* hb3      = (const float*)d_in[30];
    float* out = (float*)d_out;

    float* ws    = (float*)d_ws;
    float* bufH  = ws;                    // N x 256
    float* bufA  = bufH  + NC;            // N x 256 (agg / hl / mlp out)
    float* bufT2 = bufA  + NC;            // N x 256 (z2+h / o / out)
    float* bufT1 = bufT2 + NC;            // N x 512 (gine hidden / ha / mlp hid)
    float* bufG  = bufT1 + 2 * NC;        // 512 x 256
    float* bufG1 = bufG  + (size_t)Bg * 256;  // 512 x 128
    float* bufG2 = bufG1 + (size_t)Bg * 128;  // 512 x 64
    float* stats = bufG2 + (size_t)Bg * 64;   // 2 x 256

    const int nBlkNC = (int)(NC / 256);   // 32768

    // ---- input stage: BN(pe), build h ----
    zero_kernel<<<2, 256, 0, stream>>>(stats, 512);
    bnstats_kernel<<<256, 256, 0, stream>>>(pe, N, 20, stats, 128);
    buildh_kernel<<<N, 256, 0, stream>>>(x, pe, stats, png, pnb, node_emb,
                                         plw, plb, bufH, N);

    for (int l = 0; l < L; ++l) {
        const float* ngl = ng + (size_t)l * 3 * C;
        const float* nbl = nb + (size_t)l * 3 * C;

        // GINE: agg = scatter(relu(h[src]+ea)); z = h + agg
        zero_kernel<<<nBlkNC, 256, 0, stream>>>(bufA, (int)NC);
        scatter_kernel<<<E / 4, 256, 0, stream>>>(bufH, edge_emb, ei, ea,
                                                  bufA, E);
        add_kernel<<<nBlkNC, 256, 0, stream>>>(bufA, bufH, bufA, (int)NC);
        // t1 = relu(z @ w1 + b1); t2 = t1 @ w2 + b2 + h
        gemm_kernel<true, false><<<dim3(N / 128, 4), 256, 0, stream>>>(
            bufA, gw1 + (size_t)l * C * C, gb1 + (size_t)l * C, nullptr,
            bufT1, N, C, C);
        gemm_kernel<false, true><<<dim3(N / 128, 4), 256, 0, stream>>>(
            bufT1, gw2 + (size_t)l * C * C, gb2 + (size_t)l * C, bufH,
            bufT2, N, C, C);
        // h_local = BN(t2) -> bufA
        zero_kernel<<<2, 256, 0, stream>>>(stats, 512);
        bnstats_kernel<<<256, 256, 0, stream>>>(bufT2, N, C, stats, 128);
        bnapply_kernel<<<nBlkNC, 256, 0, stream>>>(bufT2, bufA, stats,
                                                   ngl, nbl, N, C);

        // fused attention (qkv + softmax + att@v) -> o in bufT2
        attn_kernel<<<dim3(Bg, 4), 256, 0, stream>>>(
            bufH, aiw + (size_t)l * C * 768, aib + (size_t)l * 768,
            bufT2, 0.125f);
        // proj + residual h -> bufT1 (first N*256)
        gemm_kernel<false, true><<<dim3(N / 128, 4), 256, 0, stream>>>(
            bufT2, aow + (size_t)l * C * C, aob + (size_t)l * C, bufH,
            bufT1, N, C, C);
        // h_attn = BN -> in place
        zero_kernel<<<2, 256, 0, stream>>>(stats, 512);
        bnstats_kernel<<<256, 256, 0, stream>>>(bufT1, N, C, stats, 128);
        bnapply_kernel<<<nBlkNC, 256, 0, stream>>>(bufT1, bufT1, stats,
                                                   ngl + C, nbl + C, N, C);

        // out = h_local + h_attn -> bufT2; MLP with residual -> bufA
        add_kernel<<<nBlkNC, 256, 0, stream>>>(bufA, bufT1, bufT2, (int)NC);
        gemm_kernel<true, false><<<dim3(N / 128, 8), 256, 0, stream>>>(
            bufT2, m1w + (size_t)l * C * 512, m1b + (size_t)l * 512, nullptr,
            bufT1, N, 512, C);
        gemm_kernel<false, true><<<dim3(N / 128, 4), 256, 0, stream>>>(
            bufT1, m2w + (size_t)l * 512 * C, m2b + (size_t)l * C, bufT2,
            bufA, N, C, 512);
        // h = BN(out) -> bufH
        zero_kernel<<<2, 256, 0, stream>>>(stats, 512);
        bnstats_kernel<<<256, 256, 0, stream>>>(bufA, N, C, stats, 128);
        bnapply_kernel<<<nBlkNC, 256, 0, stream>>>(bufA, bufH, stats,
                                                   ngl + 2 * C, nbl + 2 * C,
                                                   N, C);
    }

    // ---- readout head ----
    gsum_kernel<<<Bg, 256, 0, stream>>>(bufH, bufG, NPG, C);
    gemm_kernel<true, false><<<dim3(Bg / 128, 2), 256, 0, stream>>>(
        bufG, hw1, hb1, nullptr, bufG1, Bg, 128, 256);
    gemm_kernel<true, false><<<dim3(Bg / 128, 1), 256, 0, stream>>>(
        bufG1, hw2, hb2, nullptr, bufG2, Bg, 64, 128);
    headfinal_kernel<<<2, 256, 0, stream>>>(bufG2, hw3, hb3, out, Bg, 64);
}
